// DeepTraceLayer_27504970564015
// MI455X (gfx1250) — compile-verified, matrix operand order checked by reading, not run on hardware
//
#include <hip/hip_runtime.h>

// ---------------------------------------------------------------------------
// DeepTraceLayer on MI455X (gfx1250): fused graph-LSTM via v_wmma_f32_16x16x32_bf16
//   N=20000 nodes, D=H=128, DEG=32 neighbors, gates 4H=512, K_cat = D+H = 256
// LDS-resident swizzled bf16 weights, depth-3 software-pipelined B-fragment
// fetch. Weight-window base pointers are built from folder-opaque offsets so
// every DS access keeps a 16-bit immediate and no VALU address math lands on
// just-consumed WMMA operand registers (avoids the WAR-hazard v_nop bursts).
// ---------------------------------------------------------------------------

typedef __attribute__((ext_vector_type(16))) __bf16 v16bf;
typedef __attribute__((ext_vector_type(2)))  __bf16 v2bf;
typedef __attribute__((ext_vector_type(8)))  float  v8f;

#define N_NODES    20000
#define DEG        32
#define DIM        128          // D == H == 128
#define WAVES      8            // waves per workgroup
#define NODE_TILE  16           // nodes per wave
#define WCAT_TILES 256          // 32 ntiles (512/16) x 8 ktiles (256/32)
#define WOUT_TILES 64           // 8 ntiles (128/16) x 8 ktiles (256/32)
#define WCAT_U16   (WCAT_TILES * 32 * 16)       // 131072 u16 = 256 KB
#define LDS_W_BYTES (WCAT_U16 * 2)              // 262144 B
#define HBUF_U16   (NODE_TILE * DIM)            // 2048 u16 per wave
#define LDS_HB_BYTES (WAVES * HBUF_U16 * 2)     // 32768 B
#define LDS_BIAS_BYTES 2048                     // 512 f32 combined gate bias
#define LDS_TOTAL  (LDS_W_BYTES + LDS_HB_BYTES + LDS_BIAS_BYTES)   // 296960 < 320 KB

union Frag {
    v16bf          v;
    unsigned short u[16];
    unsigned int   d[8];
    uint4          q[2];
};

__device__ __forceinline__ unsigned short f2bf(float f) {
    unsigned int u = __float_as_uint(f);
    unsigned int r = u + 0x7FFFu + ((u >> 16) & 1u);   // round-to-nearest-even
    return (unsigned short)(r >> 16);
}

// Native bf16 convert: backend selects v_cvt_pk_bf16_f32 on gfx1250.
__device__ __forceinline__ unsigned int pack_bf2(float x, float y) {
    v2bf r = {(__bf16)x, (__bf16)y};
    unsigned int u;
    __builtin_memcpy(&u, &r, 4);
    return u;
}

// Constant the instruction-offset folder cannot see through: forces the
// compiler to keep (base + imm) as a live VGPR instead of re-materializing
// per-load v_add's (which would WAR-hazard against in-flight WMMA operands).
__device__ __forceinline__ unsigned opaque_imm(unsigned imm) {
    unsigned o;
    asm("v_mov_b32 %0, %1" : "=v"(o) : "i"(imm));
    return o;
}

__device__ __forceinline__ float fast_rcp(float x) {
#if __has_builtin(__builtin_amdgcn_rcpf)
    return __builtin_amdgcn_rcpf(x);
#else
    return 1.0f / x;
#endif
}

__device__ __forceinline__ float fast_sigmoid(float x) {
    return fast_rcp(1.0f + __expf(-x));
}

__device__ __forceinline__ float fast_tanh(float x) {
#if __has_builtin(__builtin_amdgcn_tanhf)
    return __builtin_amdgcn_tanhf(x);            // v_tanh_f32
#elif __has_builtin(__builtin_amdgcn_tanh_f32)
    return __builtin_amdgcn_tanh_f32(x);
#else
    return 1.0f - 2.0f * fast_rcp(__expf(2.0f * x) + 1.0f);
#endif
}

// B-fragment fetch from LDS-resident swizzled W_cat.
// Flattened tile index g in [0,256): hc = g>>5, q = (g>>3)&3, kt = g&7.
// Byte offset = q*65536 + hc*8192 + kt*1024 (+ lane*32 folded into base).
// wq[q] bases are opaque, so hc*8192 + kt*1024 + {0,16} <= 64528 stays a
// 16-bit DS immediate on every load.
__device__ __forceinline__ void loadB(Frag& b, const char* const wq[4], int g) {
    int hc = g >> 5, q = (g >> 3) & 3, kt = g & 7;
    const uint4* p = (const uint4*)(wq[q] + hc * 8192 + kt * 1024);
    b.q[0] = p[0];
    b.q[1] = p[1];
}

// ---------------------------------------------------------------------------
// Kernel 1: fp32 weights -> bf16, pre-swizzled per-lane WMMA B-fragments.
// Region 0: W_cat = [W_ih | W_hh] as 32x8 tiles (256 KB).
// Region 1 (@ WCAT_U16): W_out as 8x8 tiles (64 KB).
// Fragment element e of lane L (half=L>>4): K = kt*32 + 2*(e&8) + half*8 + (e&7).
// ---------------------------------------------------------------------------
__global__ void prep_weights_kernel(const float* __restrict__ W_ih,
                                    const float* __restrict__ W_hh,
                                    const float* __restrict__ W_out,
                                    unsigned short* __restrict__ wt) {
    int tid  = blockIdx.x * blockDim.x + threadIdx.x;
    int lane = tid & 31;
    int item = tid >> 5;
    int c    = lane & 15;
    int half = lane >> 4;

    if (item < WCAT_TILES) {
        int nt = item >> 3, kt = item & 7;
        int g  = nt * 16 + c;              // gate output column (0..511)
        #pragma unroll
        for (int e = 0; e < 16; ++e) {
            int K = kt * 32 + ((e & 8) << 1) + half * 8 + (e & 7);   // 0..255
            float v = (K < DIM) ? W_ih[g * DIM + K] : W_hh[g * DIM + (K - DIM)];
            wt[(item * 32 + lane) * 16 + e] = f2bf(v);
        }
    } else if (item < WCAT_TILES + WOUT_TILES) {
        int it = item - WCAT_TILES;
        int nt = it >> 3, kt = it & 7;
        int g  = nt * 16 + c;              // output column (0..127)
        #pragma unroll
        for (int e = 0; e < 16; ++e) {
            int K = kt * 32 + ((e & 8) << 1) + half * 8 + (e & 7);
            wt[WCAT_U16 + (it * 32 + lane) * 16 + e] = f2bf(W_out[g * 256 + K]);
        }
    }
}

// ---------------------------------------------------------------------------
// Kernel 2: fused gather + 32-step LSTM + output projection.
// 8 waves / workgroup share W_cat in LDS; each wave owns 16 nodes.
// ---------------------------------------------------------------------------
__global__ __launch_bounds__(256, 1)
void lstm_fused_kernel(const float* __restrict__ h,
                       const int*   __restrict__ nbr,
                       const float* __restrict__ b_ih,
                       const float* __restrict__ b_hh,
                       const float* __restrict__ b_out,
                       const unsigned short* __restrict__ wt,
                       float* __restrict__ out) {
    extern __shared__ char smem[];
    uint4* wlds = (uint4*)smem;                                   // 256 KB swizzled W_cat
    unsigned short* hbuf_all = (unsigned short*)(smem + LDS_W_BYTES);
    float* lbias = (float*)(smem + LDS_W_BYTES + LDS_HB_BYTES);   // combined gate bias

    // Cooperative stage of swizzled W_cat tiles + combined bias into LDS.
    const uint4* wsrc = (const uint4*)wt;
    for (int i = threadIdx.x; i < LDS_W_BYTES / 16; i += 256) wlds[i] = wsrc[i];
    for (int i = threadIdx.x; i < 512; i += 256) lbias[i] = b_ih[i] + b_hh[i];
    __syncthreads();

    int lane = threadIdx.x & 31;
    int wave = threadIdx.x >> 5;
    int tile = blockIdx.x * WAVES + wave;                         // 16-node tile id
    if (tile * NODE_TILE >= N_NODES) return;                      // whole-wave skip

    int row  = lane & 15;                 // A-operand row / C-tile column
    int half = lane >> 4;
    int node = tile * NODE_TILE + row;

    // Per-lane, per-gate base pointers into the weight tiles (64 KB windows).
    // Offsets are opaque so each base stays a distinct live VGPR.
    const char* wl = smem + lane * 32;
    const char* wq[4];
    wq[0] = wl;
    wq[1] = wl + opaque_imm(65536);
    wq[2] = wl + opaque_imm(131072);
    wq[3] = wl + opaque_imm(196608);

    unsigned short* hbuf = hbuf_all + wave * HBUF_U16;            // [16 nodes][128] bf16
    uint4* hb4 = (uint4*)hbuf;
    const uint4 z4 = {0u, 0u, 0u, 0u};
    for (int i = lane; i < HBUF_U16 / 8; i += 32) hb4[i] = z4;    // h0 = 0
    const uint4* hbq = (const uint4*)hbuf;

    // Cell state: c[16 nodes x 128 hidden] in registers, C-tile layout.
    v8f cst[8];
    #pragma unroll
    for (int hc = 0; hc < 8; ++hc)
        #pragma unroll
        for (int v = 0; v < 8; ++v) cst[hc][v] = 0.0f;

    for (int t = 0; t < DEG; ++t) {
        // --- h_prev A-fragments from LDS (2x ds_load_b128 each) ---
        Frag hf[4];
        #pragma unroll
        for (int kt = 0; kt < 4; ++kt) {
            hf[kt].q[0] = hbq[row * 16 + kt * 4 + half];
            hf[kt].q[1] = hbq[row * 16 + kt * 4 + half + 2];
        }

        // --- gathered neighbor A-fragments (4x global_load_b128 each) ---
        int idx = nbr[node * DEG + t];
        const float4* xr4 = (const float4*)(h + (long long)idx * DIM);
        Frag xf[4];
        #pragma unroll
        for (int kt = 0; kt < 4; ++kt) {
            float4 f0 = xr4[kt * 8 + half * 2];
            float4 f1 = xr4[kt * 8 + half * 2 + 1];
            float4 f2 = xr4[kt * 8 + half * 2 + 4];
            float4 f3 = xr4[kt * 8 + half * 2 + 5];
            xf[kt].d[0] = pack_bf2(f0.x, f0.y); xf[kt].d[1] = pack_bf2(f0.z, f0.w);
            xf[kt].d[2] = pack_bf2(f1.x, f1.y); xf[kt].d[3] = pack_bf2(f1.z, f1.w);
            xf[kt].d[4] = pack_bf2(f2.x, f2.y); xf[kt].d[5] = pack_bf2(f2.z, f2.w);
            xf[kt].d[6] = pack_bf2(f3.x, f3.y); xf[kt].d[7] = pack_bf2(f3.z, f3.w);
        }

        // --- software-pipelined B fetch (depth 3) over all 256 tiles ---
        Frag bb[4];
        loadB(bb[0], wq, 0);
        loadB(bb[1], wq, 1);
        loadB(bb[2], wq, 2);

        #pragma unroll
        for (int hc = 0; hc < 8; ++hc) {
            // Gate biases from LDS: issued here, only consumed in the cell
            // math below, so they never stall the B pipeline.
            float bi = lbias[      hc * 16 + row];
            float bf = lbias[128 + hc * 16 + row];
            float bg = lbias[256 + hc * 16 + row];
            float bo = lbias[384 + hc * 16 + row];

            v8f acc[4];
            #pragma unroll
            for (int q = 0; q < 4; ++q)
                #pragma unroll
                for (int v = 0; v < 8; ++v) acc[q][v] = 0.0f;

            #pragma unroll
            for (int i = 0; i < 32; ++i) {
                int g = hc * 32 + i;                 // flattened tile index
                if (g + 3 < 256) loadB(bb[(g + 3) & 3], wq, g + 3);
                int q = i >> 3, kt = i & 7;
                const Frag& a = (kt < 4) ? xf[kt] : hf[kt - 4];
                acc[q] = __builtin_amdgcn_wmma_f32_16x16x32_bf16(
                    false, a.v, false, bb[g & 3].v, (short)0, acc[q], false, false);
            }

            // --- LSTM cell ---
            #pragma unroll
            for (int v = 0; v < 8; ++v) {
                float iv = fast_sigmoid(acc[0][v] + bi);
                float fv = fast_sigmoid(acc[1][v] + bf);
                float gv = fast_tanh(acc[2][v] + bg);
                float ov = fast_sigmoid(acc[3][v] + bo);
                float c  = fv * cst[hc][v] + iv * gv;
                cst[hc][v] = c;
                // C-tile element (M = v + 8*half, N = hc*16 + row) -> h buffer
                hbuf[(v + 8 * half) * DIM + hc * 16 + row] = f2bf(ov * fast_tanh(c));
            }
        }
    }

    // --- output projection: relu(concat(h_node, h_final) @ W_out^T + b_out) ---
    const float4* nr4 = (const float4*)(h + (long long)node * DIM);
    Frag nf[4], hfin[4];
    #pragma unroll
    for (int kt = 0; kt < 4; ++kt) {
        float4 f0 = nr4[kt * 8 + half * 2];
        float4 f1 = nr4[kt * 8 + half * 2 + 1];
        float4 f2 = nr4[kt * 8 + half * 2 + 4];
        float4 f3 = nr4[kt * 8 + half * 2 + 5];
        nf[kt].d[0] = pack_bf2(f0.x, f0.y); nf[kt].d[1] = pack_bf2(f0.z, f0.w);
        nf[kt].d[2] = pack_bf2(f1.x, f1.y); nf[kt].d[3] = pack_bf2(f1.z, f1.w);
        nf[kt].d[4] = pack_bf2(f2.x, f2.y); nf[kt].d[5] = pack_bf2(f2.z, f2.w);
        nf[kt].d[6] = pack_bf2(f3.x, f3.y); nf[kt].d[7] = pack_bf2(f3.z, f3.w);
        hfin[kt].q[0] = hbq[row * 16 + kt * 4 + half];
        hfin[kt].q[1] = hbq[row * 16 + kt * 4 + half + 2];
    }

    const uint4* woutq = (const uint4*)(wt + WCAT_U16);
    #pragma unroll
    for (int nt = 0; nt < 8; ++nt) {
        v8f acc;
        float bb = b_out[nt * 16 + row];
        #pragma unroll
        for (int v = 0; v < 8; ++v) acc[v] = bb;
        #pragma unroll
        for (int kt = 0; kt < 8; ++kt) {
            Frag b;
            const uint4* bp = woutq + (unsigned)((nt * 8 + kt) * 32 + lane) * 2u;
            b.q[0] = bp[0];
            b.q[1] = bp[1];
            const Frag& a = (kt < 4) ? nf[kt] : hfin[kt - 4];
            acc = __builtin_amdgcn_wmma_f32_16x16x32_bf16(
                false, a.v, false, b.v, (short)0, acc, false, false);
        }
        #pragma unroll
        for (int v = 0; v < 8; ++v) {
            float o = acc[v];
            out[(long long)(tile * NODE_TILE + v + 8 * half) * DIM + nt * 16 + row] =
                (o > 0.0f) ? o : 0.0f;
        }
    }
}

// ---------------------------------------------------------------------------
// Host launcher. Inputs (setup_inputs order):
//   0:h [N,128] f32   1:nbr_idx [N,32] i32   2:W_ih [512,128] f32
//   3:W_hh [512,128]  4:b_ih [512]           5:b_hh [512]
//   6:W_out [128,256] 7:b_out [128]          out: [N,128] f32
// ---------------------------------------------------------------------------
extern "C" void kernel_launch(void* const* d_in, const int* in_sizes, int n_in,
                              void* d_out, int out_size, void* d_ws, size_t ws_size,
                              hipStream_t stream) {
    (void)in_sizes; (void)n_in; (void)out_size; (void)ws_size;
    const float* h     = (const float*)d_in[0];
    const int*   nbr   = (const int*)  d_in[1];
    const float* W_ih  = (const float*)d_in[2];
    const float* W_hh  = (const float*)d_in[3];
    const float* b_ih  = (const float*)d_in[4];
    const float* b_hh  = (const float*)d_in[5];
    const float* W_out = (const float*)d_in[6];
    const float* b_out = (const float*)d_in[7];
    unsigned short* wt = (unsigned short*)d_ws;   // 320 KB of swizzled bf16 weights
    float* out = (float*)d_out;

    // Allow ~290 KB dynamic LDS (gfx1250 WGP supports 320 KB).
    (void)hipFuncSetAttribute((const void*)lstm_fused_kernel,
                              hipFuncAttributeMaxDynamicSharedMemorySize, LDS_TOTAL);

    // 320 tiles x 32 lanes = 10240 threads
    prep_weights_kernel<<<40, 256, 0, stream>>>(W_ih, W_hh, W_out, wt);

    int n_tiles = (N_NODES + NODE_TILE - 1) / NODE_TILE;        // 1250
    int n_wgs   = (n_tiles + WAVES - 1) / WAVES;                // 157
    lstm_fused_kernel<<<n_wgs, 256, LDS_TOTAL, stream>>>(
        h, nbr, b_ih, b_hh, b_out, wt, out);
}
